// WaveRNN_16277926052394
// MI455X (gfx1250) — compile-verified, matrix operand order checked by reading, not run on hardware
//
#include <hip/hip_runtime.h>
#include <hip/hip_bf16.h>
#include <math.h>

// ---------------------------------------------------------------------------
// WaveRNN for MI455X (gfx1250, wave32, WMMA)
//   - bf16 WMMA (v_wmma_f32_16x16x32_bf16) for the recurrent matmul + heads
//   - persistent 28-WG scan kernel, 1 global barrier per timestep
//   - input projection fused into the scan (no 352MB rnn_input buffer)
// ---------------------------------------------------------------------------

typedef __bf16 bf16;
typedef __bf16 v16bf __attribute__((ext_vector_type(16)));
typedef __bf16 v8bf  __attribute__((ext_vector_type(8)));
typedef float  v8f   __attribute__((ext_vector_type(8)));

#define HID   896
#define HALF  448
#define G3    (3 * HID)        // 2688
#define NB    16               // batch
#define NT    2048             // timesteps
#define NBINS 256
#define NWG   28               // scan workgroups (28 * 32 == HID)
#define CPW   32               // hidden columns per scan workgroup
#define KT    (HID / 32)       // 28 k-tiles for the recurrent matmul

// ---- WMMA helpers ----------------------------------------------------------

__device__ __forceinline__ v16bf load_tile16(const bf16* p) {
  // p already includes row*stride + (lane>>4)*8; ISA 16-bit A/B layout:
  // VGPR0..3 = k in [8*half, 8*half+8), VGPR4..7 = k in [16+8*half, 16+8*half+8)
  v8bf lo = *(const v8bf*)(p);
  v8bf hi = *(const v8bf*)(p + 16);
  return __builtin_shufflevector(lo, hi, 0,1,2,3,4,5,6,7,8,9,10,11,12,13,14,15);
}

__device__ __forceinline__ v8f wmma_bf16(v16bf a, v16bf b, v8f c) {
  return __builtin_amdgcn_wmma_f32_16x16x32_bf16(
      /*neg_a=*/false, a, /*neg_b=*/false, b,
      /*c_mod=*/(short)0, c, /*reuse_a=*/false, /*reuse_b=*/false);
}

__device__ __forceinline__ float scale_sig(int v) {
  return (float)v * (2.0f / (float)(NBINS - 1)) - 1.0f;
}

// ---- one-time prep ---------------------------------------------------------

// dst[n*K + k] = bf16(src[k*N + n])   (transpose + convert, Wt is [N][K])
__global__ void k_cvt_transpose(const float* __restrict__ src,
                                bf16* __restrict__ dst, int K, int N) {
  int i = blockIdx.x * 256 + threadIdx.x;
  if (i >= K * N) return;
  int n = i / K, k = i % K;
  dst[i] = (bf16)src[k * N + n];
}

__global__ void k_init(float* __restrict__ hf, bf16* __restrict__ hb,
                       unsigned int* __restrict__ cnt) {
  int i = blockIdx.x * 256 + threadIdx.x;
  if (i == 0) *cnt = 0u;
  if (i < NB * HID) { hf[i] = 0.0f; hb[i] = (bf16)0.0f; }
}

// ---- persistent GRU scan ---------------------------------------------------

__global__ void __launch_bounds__(192)
k_scan(const bf16* __restrict__ whh_t,      // [2688][896] bf16 (n-major)
       float* __restrict__ hf32,            // [16][896] state f32
       bf16*  __restrict__ hb16,            // [16][896] state bf16 (A matrix)
       bf16*  __restrict__ hist,            // [16][2048][896] bf16
       float* __restrict__ lasth,           // [16][896] -> d_out tail
       const float* __restrict__ cond,      // [16][2048][3][896]
       const int*   __restrict__ isig,      // [16][2048][2]
       const int*   __restrict__ tgtc,      // [16][2048]
       const float* __restrict__ Wc,        // [2][1344]
       const float* __restrict__ Wf,        // [3][1344]
       const float* __restrict__ b_ih,      // [2688]
       const float* __restrict__ b_hh,      // [2688]
       unsigned int* __restrict__ cnt) {
  const int wg   = blockIdx.x;            // 0..27
  const int tid  = threadIdx.x;           // 0..191
  const int wave = tid >> 5;              // 0..5
  const int lane = tid & 31;
  const int lrow = lane & 15, lhalf = lane >> 4;
  const int g    = wave >> 1;             // gate 0..2
  const int ntl  = wave & 1;              // 0..1 (16-col tile within WG)
  const int colbase = wg * CPW;

  __shared__ float gbuf[3][16 * CPW];     // gate pre-activations (h @ W_hh part)

  // Lane-adjusted tile base pointers (WMMA A/B 16-bit layout).
  const bf16* Ap = hb16 + (size_t)lrow * HID + lhalf * 8;
  const bf16* Bp = whh_t +
      ((size_t)(g * HID + colbase + ntl * 16 + lrow)) * HID + lhalf * 8;

  // --- per-thread elementwise slots (512 elems / 192 threads -> <=3 each) ---
  struct Slot {
    int m, col, valid;
    float bih[3], bhh[3], c0[3], c1[3], c2[3];
  } sl[3];
#pragma unroll
  for (int s = 0; s < 3; ++s) {
    int e = tid + s * 192;
    sl[s].valid = (e < NB * CPW);
    if (sl[s].valid) {
      int m = e >> 5, c = e & 31, col = colbase + c;
      sl[s].m = m; sl[s].col = col;
#pragma unroll
      for (int gg = 0; gg < 3; ++gg) {
        sl[s].bih[gg] = b_ih[gg * HID + col];
        sl[s].bhh[gg] = b_hh[gg * HID + col];
        if (col < HALF) {
          sl[s].c0[gg] = Wc[0 * (3 * HALF) + gg * HALF + col];
          sl[s].c1[gg] = Wc[1 * (3 * HALF) + gg * HALF + col];
          sl[s].c2[gg] = 0.0f;
        } else {
          int j = col - HALF;
          sl[s].c0[gg] = Wf[0 * (3 * HALF) + gg * HALF + j];
          sl[s].c1[gg] = Wf[1 * (3 * HALF) + gg * HALF + j];
          sl[s].c2[gg] = Wf[2 * (3 * HALF) + gg * HALF + j];
        }
      }
    }
  }

  for (int t = 0; t < NT; ++t) {
    // ---- h_t @ W_hh for this wave's 16 columns of gate g -------------------
    v8f acc0 = {}; v8f acc1 = {};
#pragma unroll 4
    for (int ki = 0; ki < KT; ki += 2) {
      v16bf a0 = load_tile16(Ap + ki * 32);
      v16bf b0 = load_tile16(Bp + ki * 32);
      acc0 = wmma_bf16(a0, b0, acc0);
      v16bf a1 = load_tile16(Ap + (ki + 1) * 32);
      v16bf b1 = load_tile16(Bp + (ki + 1) * 32);
      acc1 = wmma_bf16(a1, b1, acc1);
    }
    // C layout: VGPR r, lanes 0-15 -> M=r, lanes 16-31 -> M=8+r; N = lane%16
#pragma unroll
    for (int r = 0; r < 8; ++r) {
      gbuf[g][(lhalf * 8 + r) * CPW + ntl * 16 + lrow] = acc0[r] + acc1[r];
    }
    __syncthreads();

    // ---- fused input projection + GRU elementwise --------------------------
#pragma unroll
    for (int s = 0; s < 3; ++s) {
      if (!sl[s].valid) continue;
      const int m = sl[s].m, col = sl[s].col, c = col - colbase;
      const size_t bt = (size_t)m * NT + t;
      const float s0 = scale_sig(isig[bt * 2 + 0]);
      const float s1 = scale_sig(isig[bt * 2 + 1]);
      const float tc = scale_sig(tgtc[bt]);
      const size_t cb = (bt * 3) * HID + col;
      float x[3], hgate[3];
#pragma unroll
      for (int gg = 0; gg < 3; ++gg) {
        x[gg] = sl[s].c0[gg] * s0 + sl[s].c1[gg] * s1 + sl[s].c2[gg] * tc +
                sl[s].bih[gg] + cond[cb + (size_t)gg * HID];
        hgate[gg] = gbuf[gg][m * CPW + c] + sl[s].bhh[gg];
      }
      const float r = 1.0f / (1.0f + expf(-(x[0] + hgate[0])));
      const float z = 1.0f / (1.0f + expf(-(x[1] + hgate[1])));
      const float n = tanhf(x[2] + r * hgate[2]);
      const float hprev = hf32[m * HID + col];
      const float hnew  = (1.0f - z) * n + z * hprev;
      hf32[m * HID + col] = hnew;
      hb16[m * HID + col] = (bf16)hnew;
      hist[bt * HID + col] = (bf16)hnew;
      if (t == NT - 1) lasth[m * HID + col] = hnew;
    }

    // ---- device-wide step barrier ------------------------------------------
    __threadfence();
    __syncthreads();
    if (tid == 0) {
      __hip_atomic_fetch_add(cnt, 1u, __ATOMIC_RELEASE, __HIP_MEMORY_SCOPE_AGENT);
      const unsigned int target = (unsigned int)NWG * (unsigned int)(t + 1);
      while (__hip_atomic_load(cnt, __ATOMIC_RELAXED, __HIP_MEMORY_SCOPE_AGENT) <
             target) {
        __builtin_amdgcn_s_sleep(1);
      }
    }
    __syncthreads();
    // acquire per wave -> invalidate caches so new h is visible
    (void)__hip_atomic_load(cnt, __ATOMIC_ACQUIRE, __HIP_MEMORY_SCOPE_AGENT);
  }
}

// ---- generic WMMA GEMM: C[M,N] = act(A[M,K]bf16 @ Bt[N,K]bf16 + bias) ------

__global__ void __launch_bounds__(32)
k_gemm(const bf16* __restrict__ A, int ldA,
       const bf16* __restrict__ Bt, int K,
       const float* __restrict__ bias,
       float* __restrict__ Cf, bf16* __restrict__ Cb, int ldC, int relu) {
  const int mt = blockIdx.x, nt = blockIdx.y;
  const int lane = threadIdx.x;
  const int lrow = lane & 15, lhalf = lane >> 4;
  const bf16* Ap = A + (size_t)(mt * 16 + lrow) * ldA + lhalf * 8;
  const bf16* Bp = Bt + (size_t)(nt * 16 + lrow) * K + lhalf * 8;

  v8f acc = {};
  for (int k0 = 0; k0 < K; k0 += 32) {
    __builtin_prefetch(Bp + k0 + 128, 0, 1);   // global_prefetch_b8
    v16bf a = load_tile16(Ap + k0);
    v16bf b = load_tile16(Bp + k0);
    acc = wmma_bf16(a, b, acc);
  }

  const int n  = nt * 16 + lrow;
  const int m0 = mt * 16 + lhalf * 8;
  const float bv = bias[n];
#pragma unroll
  for (int r = 0; r < 8; ++r) {
    float v = acc[r] + bv;
    if (relu) v = v > 0.0f ? v : 0.0f;
    const size_t idx = (size_t)(m0 + r) * ldC + n;
    if (Cb) Cb[idx] = (bf16)v;
    else    Cf[idx] = v;
  }
}

// ---------------------------------------------------------------------------

extern "C" void kernel_launch(void* const* d_in, const int* in_sizes, int n_in,
                              void* d_out, int out_size, void* d_ws,
                              size_t ws_size, hipStream_t stream) {
  const float* cond  = (const float*)d_in[0];
  const int*   isig  = (const int*)  d_in[1];
  const int*   tgtc  = (const int*)  d_in[2];
  const float* Wc    = (const float*)d_in[3];
  const float* Wf    = (const float*)d_in[4];
  const float* Whh   = (const float*)d_in[5];
  const float* b_ih  = (const float*)d_in[6];
  const float* b_hh  = (const float*)d_in[7];
  const float* W1c   = (const float*)d_in[8];
  const float* b1c   = (const float*)d_in[9];
  const float* W2c   = (const float*)d_in[10];
  const float* b2c   = (const float*)d_in[11];
  const float* W1f   = (const float*)d_in[12];
  const float* b1f   = (const float*)d_in[13];
  const float* W2f   = (const float*)d_in[14];
  const float* b2f   = (const float*)d_in[15];

  // ---- workspace layout (256B aligned blocks, ~124 MB total) ----
  char* ws = (char*)d_ws;
  size_t off = 0;
  auto alloc = [&](size_t bytes) {
    size_t o = off;
    off += (bytes + 255) & ~(size_t)255;
    return o;
  };
  unsigned int* cnt = (unsigned int*)(ws + alloc(4));
  float* hf32  = (float*)(ws + alloc((size_t)NB * HID * 4));
  bf16*  hb16  = (bf16*) (ws + alloc((size_t)NB * HID * 2));
  bf16*  whh_t = (bf16*) (ws + alloc((size_t)G3 * HID * 2));
  bf16*  w1c_t = (bf16*) (ws + alloc((size_t)HALF * HALF * 2));
  bf16*  w2c_t = (bf16*) (ws + alloc((size_t)NBINS * HALF * 2));
  bf16*  w1f_t = (bf16*) (ws + alloc((size_t)HALF * HALF * 2));
  bf16*  w2f_t = (bf16*) (ws + alloc((size_t)NBINS * HALF * 2));
  bf16*  hist  = (bf16*) (ws + alloc((size_t)NB * NT * HID * 2));
  bf16*  act_c = (bf16*) (ws + alloc((size_t)NB * NT * HALF * 2));
  bf16*  act_f = (bf16*) (ws + alloc((size_t)NB * NT * HALF * 2));

  float* out_c = (float*)d_out;                       // [16][2048][256]
  float* out_f = out_c + (size_t)NB * NT * NBINS;     // [16][2048][256]
  float* lasth = out_f + (size_t)NB * NT * NBINS;     // [16][896]

  // ---- prep: state init + weight transpose/convert to bf16 ----
  k_init<<<(NB * HID + 255) / 256, 256, 0, stream>>>(hf32, hb16, cnt);
  k_cvt_transpose<<<(HID * G3 + 255) / 256, 256, 0, stream>>>(Whh, whh_t, HID, G3);
  k_cvt_transpose<<<(HALF * HALF + 255) / 256, 256, 0, stream>>>(W1c, w1c_t, HALF, HALF);
  k_cvt_transpose<<<(HALF * NBINS + 255) / 256, 256, 0, stream>>>(W2c, w2c_t, HALF, NBINS);
  k_cvt_transpose<<<(HALF * HALF + 255) / 256, 256, 0, stream>>>(W1f, w1f_t, HALF, HALF);
  k_cvt_transpose<<<(HALF * NBINS + 255) / 256, 256, 0, stream>>>(W2f, w2f_t, HALF, NBINS);

  // ---- sequential GRU scan (persistent, one barrier per step) ----
  k_scan<<<NWG, 192, 0, stream>>>(whh_t, hf32, hb16, hist, lasth, cond, isig,
                                  tgtc, Wc, Wf, b_ih, b_hh, cnt);

  // ---- output heads: relu(hc @ W1 + b1) @ W2 + b2 ----
  const int MT = (NB * NT) / 16;                      // 2048 row tiles
  dim3 g1(MT, HALF / 16);                             // 2048 x 28
  k_gemm<<<g1, 32, 0, stream>>>(hist, HID, w1c_t, HALF, b1c,
                                nullptr, act_c, HALF, 1);
  k_gemm<<<g1, 32, 0, stream>>>(hist, HID, w1f_t, HALF, b1f,
                                nullptr, act_f, HALF, 1);
  dim3 g2(MT, NBINS / 16);                            // 2048 x 16
  k_gemm<<<g2, 32, 0, stream>>>(act_c, HALF, w2c_t, HALF, b2c,
                                out_c, nullptr, NBINS, 0);
  k_gemm<<<g2, 32, 0, stream>>>(act_f, HALF, w2f_t, HALF, b2f,
                                out_f, nullptr, NBINS, 0);
  (void)in_sizes; (void)n_in; (void)out_size; (void)ws_size;
}
